// VisionTransformerMoE_3453153706568
// MI455X (gfx1250) — compile-verified, matrix operand order checked by reading, not run on hardware
//
#include <hip/hip_runtime.h>

// ---------------- problem constants ----------------
#define DMODEL   768
#define NEXP     8
#define DFF      3072
#define NCLS     1000
#define NCLS_PAD 1008     // padded to multiple of 16 for WMMA N tiles
#define BATCH    16
#define SEQ      196      // 14x14 patches
#define TOKENS   3136     // BATCH*SEQ, multiple of 16
#define CAP      490      // int(1.25 * 3136 / 8)
#define CAPP     512      // capacity padded to multiple of 16
#define IMG      224
#define PATCH    16

// GEMM block tiling for the expert FFN kernels
#define BM 128            // block tile M (8 waves x 16 rows)
#define BN 64             // block tile N (4 WMMA tiles, staged in LDS)
#define KCHUNK 256        // K halves staged per LDS round
#define KPAD   264        // padded LDS row stride (528B = 4 dwords mod 64 banks -> conflict-free)

// ---------------- WMMA / vector types ----------------
typedef __attribute__((ext_vector_type(16))) _Float16 v16h;
typedef __attribute__((ext_vector_type(8)))  _Float16 v8h;
typedef __attribute__((ext_vector_type(8)))  float    v8f;
typedef __attribute__((ext_vector_type(4)))  float    v4f;
typedef __attribute__((ext_vector_type(4)))  unsigned int u32x4;
typedef __attribute__((ext_vector_type(8)))  int      i32x8;
typedef __attribute__((ext_vector_type(4)))  int      i32x4;

union Frag16 { v16h v; v8h h[2]; };

// Convert 8 contiguous f32 -> v8h
__device__ __forceinline__ v8h cvt8_f32(const float* p) {
  v4f a = *(const v4f*)p;
  v4f b = *(const v4f*)(p + 4);
  v8h r;
  r[0]=(_Float16)a[0]; r[1]=(_Float16)a[1]; r[2]=(_Float16)a[2]; r[3]=(_Float16)a[3];
  r[4]=(_Float16)b[0]; r[5]=(_Float16)b[1]; r[6]=(_Float16)b[2]; r[7]=(_Float16)b[3];
  return r;
}

// ---------------- CDNA5 per-lane async global->LDS staging (ASYNCcnt) ----------------
// Copies BN x KCHUNK halves of the row-major weight panel into LDS with the
// padded KPAD row stride. gsrc points at B[n0][k0]; row stride = K halves.
__device__ __forceinline__ void async_stage_b(const _Float16* __restrict__ gsrc,
                                              int K, _Float16* sB) {
  const int tid = threadIdx.x;
  const unsigned int lds0 = (unsigned int)(unsigned long long)sB;
#pragma unroll
  for (int i = 0; i < (BN * KCHUNK) / (256 * 8); ++i) {   // 8 chunks of 16B / thread
    int c   = tid + i * 256;         // 16-byte chunk index, 2048 total
    int row = c >> 5;                // KCHUNK halves = 32 chunks per row
    int col = c & 31;
    unsigned long long ga =
        (unsigned long long)(const void*)(gsrc + (size_t)row * K + col * 8);
    unsigned int la = lds0 + (unsigned int)(row * KPAD + col * 8) * 2u;
    asm volatile("global_load_async_to_lds_b128 %0, %1, off"
                 :: "v"(la), "v"(ga) : "memory");
  }
  asm volatile("s_wait_asynccnt 0x0" ::: "memory");
}

// ---------------- CDNA5 Tensor Data Mover staging (TENSORcnt) ----------------
// One TDM descriptor moves the whole BN x KCHUNK panel (row-major, 2B elems)
// into LDS, inserting 4 DWORDs of padding after every 128 DWORDs (one row)
// so LDS rows land on the conflict-free KPAD stride. Wave 0 issues the DMA
// (EXEC is ignored for tensor ops) and waits on TENSORcnt; the workgroup
// barrier in the caller publishes the panel.
__device__ __forceinline__ void tdm_stage_b(const _Float16* __restrict__ gsrc,
                                            int K, int Nrows, _Float16* sB) {
  if (threadIdx.x < 32) {
    unsigned long long ga = (unsigned long long)(const void*)gsrc;
    unsigned int lds_addr = (unsigned int)(unsigned long long)sB;
    u32x4 g0;
    g0[0] = 1u;                                           // count=1, valid, user mode
    g0[1] = lds_addr;                                     // LDS dest byte address
    g0[2] = (unsigned int)(ga & 0xffffffffull);           // global_addr[31:0]
    g0[3] = (unsigned int)((ga >> 32) & 0x01ffffffull)    // global_addr[56:32]
          | 0x80000000u;                                  // type=2 ("image")
    i32x8 g1;
    g1[0] = (1 << 16)                                     // data_size: 2 bytes
          | (1 << 20)                                     // pad_enable
          | (6 << 22)                                     // pad_interval: 128 DWORDs
          | (3 << 25);                                    // pad_amount: 4 DWORDs
    g1[1] = (K & 0xffff) << 16;                           // tensor_dim0 lo16
    g1[2] = (int)(((unsigned)K >> 16) | (((unsigned)Nrows & 0xffffu) << 16));
    g1[3] = (int)(((unsigned)Nrows >> 16) | ((unsigned)KCHUNK << 16)); // tile_dim0
    g1[4] = BN;                                           // tile_dim1=64, tile_dim2=0
    g1[5] = K;                                            // tensor_dim0_stride lo32
    g1[6] = 0;
    g1[7] = 0;
    i32x4 gz = {0, 0, 0, 0};
#if defined(__clang_major__) && (__clang_major__ >= 23)
    i32x8 z8 = {0, 0, 0, 0, 0, 0, 0, 0};
    __builtin_amdgcn_tensor_load_to_lds(g0, g1, gz, gz, z8, 0);
#else
    __builtin_amdgcn_tensor_load_to_lds(g0, g1, gz, gz, 0);
#endif
    __builtin_amdgcn_s_wait_tensorcnt(0);
  }
}

// ---------------- prep kernels ----------------
__global__ void cvt_f32_to_f16_kernel(const float* __restrict__ s,
                                      _Float16* __restrict__ d, int n) {
  int i = blockIdx.x * blockDim.x + threadIdx.x;
  if (i < n) d[i] = (_Float16)s[i];
}

// dst[c][r] = src[r][c] (with c zero-padded up to colsPad), per batch z
__global__ void transpose_cvt_kernel(const float* __restrict__ src,
                                     _Float16* __restrict__ dst,
                                     int rows, int cols, int colsPad,
                                     size_t srcStride, size_t dstStride) {
  const float* s = src + (size_t)blockIdx.z * srcStride;
  _Float16*    d = dst + (size_t)blockIdx.z * dstStride;
  int idx = blockIdx.x * blockDim.x + threadIdx.x;
  int total = colsPad * rows;
  if (idx >= total) return;
  int c = idx / rows;
  int r = idx - c * rows;
  float v = (c < cols) ? s[(size_t)r * cols + c] : 0.0f;
  d[(size_t)c * rows + r] = (_Float16)v;
}

// ---------------- patch embed (WMMA GEMM vs conv weights) ----------------
__device__ __forceinline__ v8h load_patch8(const float* __restrict__ x,
                                           int b, int ph, int pw, int kg) {
  int c   = kg >> 8;
  int rem = kg & 255;
  int i   = rem >> 4;
  int j   = rem & 15;
  const float* px = x + ((((size_t)b * 3 + c) * IMG) + (size_t)(ph * PATCH + i)) * IMG
                      + pw * PATCH + j;
  return cvt8_f32(px);
}

__global__ void patch_embed_wmma(const float* __restrict__ x,
                                 const _Float16* __restrict__ convw, // [768][768] f16
                                 const float* __restrict__ convb,
                                 const float* __restrict__ pos,      // [5000][1][768]
                                 float* __restrict__ xt,             // [T][768] f32
                                 _Float16* __restrict__ tok) {       // [T][768] f16
  const int m0 = blockIdx.y * 16;
  const int n0 = blockIdx.x * 16;
  const int lane = threadIdx.x;
  const int mr = lane & 15;
  const int kb = (lane < 16) ? 0 : 8;

  const int t  = m0 + mr;
  const int b  = t / SEQ;
  const int s  = t - b * SEQ;
  const int ph = s / 14;
  const int pw = s - ph * 14;

  const _Float16* pb = convw + (size_t)(n0 + mr) * DMODEL;

  v8f acc = {};
  for (int k0 = 0; k0 < DMODEL; k0 += 32) {
    Frag16 fa, fb;
    fa.h[0] = load_patch8(x, b, ph, pw, k0 + kb);
    fa.h[1] = load_patch8(x, b, ph, pw, k0 + kb + 16);
    fb.h[0] = *(const v8h*)(pb + k0 + kb);
    fb.h[1] = *(const v8h*)(pb + k0 + kb + 16);
    acc = __builtin_amdgcn_wmma_f32_16x16x32_f16(false, fa.v, false, fb.v,
                                                 (short)0, acc, false, false);
  }

  const int nn = lane & 15;
  const int mb = (lane < 16) ? 0 : 8;
  const int d  = n0 + nn;
  const float cb = convb[d];
#pragma unroll
  for (int r = 0; r < 8; ++r) {
    int tt = m0 + mb + r;
    int bb = tt / SEQ;
    float v = acc[r] + cb + pos[(size_t)bb * DMODEL + d];
    xt[(size_t)tt * DMODEL + d]  = v;
    tok[(size_t)tt * DMODEL + d] = (_Float16)v;
  }
}

// ---------------- router ----------------
__global__ void router_kernel(const float* __restrict__ xt,
                              const float* __restrict__ rw,  // [768][8]
                              const float* __restrict__ rb,  // [8]
                              float* __restrict__ pmax,
                              int* __restrict__ route) {
  int gid  = blockIdx.x * blockDim.x + threadIdx.x;
  int wid  = gid >> 5;            // one wave per token
  int lane = threadIdx.x & 31;
  if (wid >= TOKENS) return;
  const float* px = xt + (size_t)wid * DMODEL;
  float acc[NEXP] = {0.f,0.f,0.f,0.f,0.f,0.f,0.f,0.f};
  for (int d = lane; d < DMODEL; d += 32) {
    float xv = px[d];
    const float* prw = rw + (size_t)d * NEXP;
#pragma unroll
    for (int e = 0; e < NEXP; ++e) acc[e] += xv * prw[e];
  }
#pragma unroll
  for (int e = 0; e < NEXP; ++e)
    for (int off = 16; off > 0; off >>= 1)
      acc[e] += __shfl_xor(acc[e], off, 32);
  if (lane == 0) {
    float l[NEXP], lmax = -3.0e38f; int am = 0;
#pragma unroll
    for (int e = 0; e < NEXP; ++e) {
      l[e] = acc[e] + rb[e];
      if (l[e] > lmax) { lmax = l[e]; am = e; }
    }
    float ssum = 0.f;
#pragma unroll
    for (int e = 0; e < NEXP; ++e) ssum += expf(l[e] - lmax);
    pmax[wid]  = 1.0f / ssum;
    route[wid] = am;
  }
}

// ---------------- capacity scan ----------------
__global__ void capacity_kernel(const int* __restrict__ route,
                                int* __restrict__ slot,
                                int* __restrict__ keep,
                                int* __restrict__ dest) { // [NEXP][CAPP], -1 = empty
  int e = threadIdx.x;
  if (e >= NEXP) return;
  int cnt = 0;
  for (int t = 0; t < TOKENS; ++t) {
    if (route[t] == e) {
      if (cnt < CAP) { slot[t] = cnt; keep[t] = 1; dest[e * CAPP + cnt] = t; }
      else           { slot[t] = 0;   keep[t] = 0; }
      ++cnt;
    }
  }
  int filled = cnt < CAP ? cnt : CAP;
  for (int c = filled; c < CAPP; ++c) dest[e * CAPP + c] = -1;
}

// ---------------- gather ----------------
__global__ void gather_kernel(const _Float16* __restrict__ tok,
                              const int* __restrict__ dest,
                              _Float16* __restrict__ xin) { // [NEXP*CAPP][768]
  int row = blockIdx.x;
  int t = dest[row];
  _Float16* o = xin + (size_t)row * DMODEL;
  if (t >= 0) {
    const _Float16* s = tok + (size_t)t * DMODEL;
    for (int d = threadIdx.x; d < DMODEL; d += blockDim.x) o[d] = s[d];
  } else {
    for (int d = threadIdx.x; d < DMODEL; d += blockDim.x) o[d] = (_Float16)0.f;
  }
}

// ---------------- shared FFN mainloop ----------------
// Block: 256 threads (8 waves). Block tile: BM x BN. Wave w owns rows
// [m0 + w*16, +16). The BN x KCHUNK weight panel lives in LDS with a
// conflict-free KPAD row stride; each B fragment is reused by all 8 waves,
// each A fragment feeds 4 WMMA. USE_TDM selects the Tensor Data Mover vs
// per-lane async-copy staging path.
template <bool USE_TDM>
__device__ __forceinline__ void ffn_mainloop(const _Float16* __restrict__ Arow,
                                             const _Float16* __restrict__ Bg,
                                             _Float16* sB, int K, int Nrows,
                                             int mr, int kb, v8f acc[4]) {
  for (int k0 = 0; k0 < K; k0 += KCHUNK) {
    if (k0) __syncthreads();                 // all waves done reading prev panel
    if (USE_TDM) tdm_stage_b(Bg + k0, K, Nrows, sB);
    else         async_stage_b(Bg + k0, K, sB);
    __syncthreads();                         // panel visible to all waves
    __builtin_prefetch((const void*)(Arow + k0 + KCHUNK), 0, 1);
#pragma unroll
    for (int ks = 0; ks < KCHUNK; ks += 32) {
      Frag16 fa;
      fa.h[0] = *(const v8h*)(Arow + k0 + ks + kb);
      fa.h[1] = *(const v8h*)(Arow + k0 + ks + kb + 16);
#pragma unroll
      for (int nt = 0; nt < 4; ++nt) {
        Frag16 fb;
        const _Float16* sp = sB + (size_t)(nt * 16 + mr) * KPAD + ks;
        fb.h[0] = *(const v8h*)(sp + kb);
        fb.h[1] = *(const v8h*)(sp + kb + 16);
        acc[nt] = __builtin_amdgcn_wmma_f32_16x16x32_f16(false, fa.v, false, fb.v,
                                                         (short)0, acc[nt],
                                                         false, false);
      }
    }
  }
}

// ---------------- expert FFN layer 1: [CAPP x 768]·[768 x 3072], relu ----------------
__global__ void ffn1_wmma(const _Float16* __restrict__ xin,   // [E][CAPP][768]
                          const _Float16* __restrict__ w1t,   // [E][3072][768]
                          const float* __restrict__ b1,       // [E][3072]
                          _Float16* __restrict__ h) {         // [E][CAPP][3072]
  __shared__ __align__(16) _Float16 sB[BN * KPAD];
  const int e  = blockIdx.z;
  const int n0 = blockIdx.x * BN;
  const int m0 = blockIdx.y * BM;
  const int wave = threadIdx.x >> 5;
  const int lane = threadIdx.x & 31;
  const int mr = lane & 15;
  const int kb = (lane < 16) ? 0 : 8;

  const _Float16* Arow = xin + ((size_t)e * CAPP + m0 + wave * 16 + mr) * DMODEL;
  const _Float16* Bg   = w1t + ((size_t)e * DFF + n0) * DMODEL;

  v8f acc[4] = {};
  ffn_mainloop<true>(Arow, Bg, sB, DMODEL, DFF, mr, kb, acc);   // TDM staging

  const int nn = lane & 15;
  const int mb = (lane < 16) ? 0 : 8;
  const int mrow0 = m0 + wave * 16 + mb;
  _Float16* H = h + (size_t)e * CAPP * DFF;
#pragma unroll
  for (int nt = 0; nt < 4; ++nt) {
    int col = n0 + nt * 16 + nn;
    float bias = b1[(size_t)e * DFF + col];
#pragma unroll
    for (int r = 0; r < 8; ++r) {
      float v = acc[nt][r] + bias;
      v = v > 0.f ? v : 0.f;                 // relu
      H[(size_t)(mrow0 + r) * DFF + col] = (_Float16)v;
    }
  }
}

// ---------------- expert FFN layer 2: [CAPP x 3072]·[3072 x 768] ----------------
__global__ void ffn2_wmma(const _Float16* __restrict__ h,     // [E][CAPP][3072]
                          const _Float16* __restrict__ w2t,   // [E][768][3072]
                          const float* __restrict__ b2,       // [E][768]
                          float* __restrict__ eo) {           // [E][CAPP][768]
  __shared__ __align__(16) _Float16 sB[BN * KPAD];
  const int e  = blockIdx.z;
  const int n0 = blockIdx.x * BN;
  const int m0 = blockIdx.y * BM;
  const int wave = threadIdx.x >> 5;
  const int lane = threadIdx.x & 31;
  const int mr = lane & 15;
  const int kb = (lane < 16) ? 0 : 8;

  const _Float16* Arow = h   + ((size_t)e * CAPP + m0 + wave * 16 + mr) * DFF;
  const _Float16* Bg   = w2t + ((size_t)e * DMODEL + n0) * DFF;

  v8f acc[4] = {};
  ffn_mainloop<false>(Arow, Bg, sB, DFF, DMODEL, mr, kb, acc); // async-copy staging

  const int nn = lane & 15;
  const int mb = (lane < 16) ? 0 : 8;
  const int mrow0 = m0 + wave * 16 + mb;
  float* O = eo + (size_t)e * CAPP * DMODEL;
#pragma unroll
  for (int nt = 0; nt < 4; ++nt) {
    int col = n0 + nt * 16 + nn;
    float bias = b2[(size_t)e * DMODEL + col];
#pragma unroll
    for (int r = 0; r < 8; ++r)
      O[(size_t)(mrow0 + r) * DMODEL + col] = acc[nt][r] + bias;
  }
}

// ---------------- scatter + passthrough + *pmax + sum pool ----------------
__global__ void scatter_pool_kernel(const float* __restrict__ eo,
                                    const float* __restrict__ xt,
                                    const int* __restrict__ route,
                                    const int* __restrict__ slot,
                                    const int* __restrict__ keep,
                                    const float* __restrict__ pmax,
                                    float* __restrict__ pooled) { // [B][768]
  int b = blockIdx.y;
  int d = blockIdx.x * blockDim.x + threadIdx.x;
  if (d >= DMODEL) return;
  float sum = 0.f;
  for (int s = 0; s < SEQ; ++s) {
    int t = b * SEQ + s;
    float v;
    if (keep[t]) {
      int e = route[t], c = slot[t];
      v = eo[((size_t)e * CAPP + c) * DMODEL + d];
    } else {
      v = xt[(size_t)t * DMODEL + d];
    }
    sum += v * pmax[t];
  }
  pooled[(size_t)b * DMODEL + d] = sum;
}

// ---------------- head: [16 x 768]·[768 x 1000] (WMMA, N guarded) ----------------
__global__ void head_wmma(const float* __restrict__ pooled,   // [16][768] f32
                          const _Float16* __restrict__ hwt,   // [1008][768] f16
                          const float* __restrict__ hb,       // [1000]
                          float* __restrict__ out) {          // [16][1000]
  const int n0 = blockIdx.x * 16;
  const int lane = threadIdx.x;
  const int mr = lane & 15;
  const int kb = (lane < 16) ? 0 : 8;
  const float*    pa = pooled + (size_t)mr * DMODEL;
  const _Float16* pb = hwt    + (size_t)(n0 + mr) * DMODEL;
  v8f acc = {};
  for (int k0 = 0; k0 < DMODEL; k0 += 32) {
    Frag16 fa, fb;
    fa.h[0] = cvt8_f32(pa + k0 + kb);
    fa.h[1] = cvt8_f32(pa + k0 + kb + 16);
    fb.h[0] = *(const v8h*)(pb + k0 + kb);
    fb.h[1] = *(const v8h*)(pb + k0 + kb + 16);
    acc = __builtin_amdgcn_wmma_f32_16x16x32_f16(false, fa.v, false, fb.v,
                                                 (short)0, acc, false, false);
  }
  const int nn = lane & 15;
  const int mb = (lane < 16) ? 0 : 8;
  const int col = n0 + nn;
  if (col < NCLS) {
    const float bias = hb[col];
#pragma unroll
    for (int r = 0; r < 8; ++r)
      out[(size_t)(mb + r) * NCLS + col] = acc[r] + bias;
  }
}

// ---------------- launch ----------------
extern "C" void kernel_launch(void* const* d_in, const int* in_sizes, int n_in,
                              void* d_out, int out_size, void* d_ws, size_t ws_size,
                              hipStream_t stream) {
  const float* x        = (const float*)d_in[0];
  const float* conv_w   = (const float*)d_in[1];
  const float* conv_b   = (const float*)d_in[2];
  const float* pos      = (const float*)d_in[3];
  const float* router_w = (const float*)d_in[4];
  const float* router_b = (const float*)d_in[5];
  const float* w1       = (const float*)d_in[6];
  const float* b1       = (const float*)d_in[7];
  const float* w2       = (const float*)d_in[8];
  const float* b2       = (const float*)d_in[9];
  const float* head_w   = (const float*)d_in[10];
  const float* head_b   = (const float*)d_in[11];
  float* out = (float*)d_out;

  // workspace carve (256B aligned)
  char* ws = (char*)d_ws;
  auto carve = [&](size_t bytes) -> char* {
    char* p = ws;
    ws += (bytes + 255) & ~(size_t)255;
    return p;
  };
  _Float16* convw_h = (_Float16*)carve((size_t)DMODEL * DMODEL * 2);
  _Float16* w1t_h   = (_Float16*)carve((size_t)NEXP * DFF * DMODEL * 2);
  _Float16* w2t_h   = (_Float16*)carve((size_t)NEXP * DMODEL * DFF * 2);
  _Float16* hwt_h   = (_Float16*)carve((size_t)NCLS_PAD * DMODEL * 2);
  _Float16* tok_h   = (_Float16*)carve((size_t)TOKENS * DMODEL * 2);
  float*    xt_f    = (float*)   carve((size_t)TOKENS * DMODEL * 4);
  float*    pmax_f  = (float*)   carve((size_t)TOKENS * 4);
  int*      route_i = (int*)     carve((size_t)TOKENS * 4);
  int*      slot_i  = (int*)     carve((size_t)TOKENS * 4);
  int*      keep_i  = (int*)     carve((size_t)TOKENS * 4);
  int*      dest_i  = (int*)     carve((size_t)NEXP * CAPP * 4);
  _Float16* xin_h   = (_Float16*)carve((size_t)NEXP * CAPP * DMODEL * 2);
  _Float16* h_h     = (_Float16*)carve((size_t)NEXP * CAPP * DFF * 2);
  float*    eo_f    = (float*)   carve((size_t)NEXP * CAPP * DMODEL * 4);
  float*    pooled  = (float*)   carve((size_t)BATCH * DMODEL * 4);

  // 1) weight prep
  {
    int n = DMODEL * DMODEL;
    cvt_f32_to_f16_kernel<<<(n + 255) / 256, 256, 0, stream>>>(conv_w, convw_h, n);
  }
  {
    int total = DFF * DMODEL;  // w1: per expert [768][3072] -> [3072][768]
    dim3 g((total + 255) / 256, 1, NEXP);
    transpose_cvt_kernel<<<g, 256, 0, stream>>>(w1, w1t_h, DMODEL, DFF, DFF,
                                                (size_t)DMODEL * DFF, (size_t)DFF * DMODEL);
  }
  {
    int total = DMODEL * DFF;  // w2: per expert [3072][768] -> [768][3072]
    dim3 g((total + 255) / 256, 1, NEXP);
    transpose_cvt_kernel<<<g, 256, 0, stream>>>(w2, w2t_h, DFF, DMODEL, DMODEL,
                                                (size_t)DFF * DMODEL, (size_t)DMODEL * DFF);
  }
  {
    int total = NCLS_PAD * DMODEL;  // head: [768][1000] -> [1008][768] zero-padded
    dim3 g((total + 255) / 256, 1, 1);
    transpose_cvt_kernel<<<g, 256, 0, stream>>>(head_w, hwt_h, DMODEL, NCLS, NCLS_PAD,
                                                0, 0);
  }

  // 2) patch embed GEMM (+bias +pos)
  {
    dim3 g(DMODEL / 16, TOKENS / 16, 1);
    patch_embed_wmma<<<g, 32, 0, stream>>>(x, convw_h, conv_b, pos, xt_f, tok_h);
  }

  // 3) router
  {
    dim3 g((TOKENS * 32 + 255) / 256, 1, 1);
    router_kernel<<<g, 256, 0, stream>>>(xt_f, router_w, router_b, pmax_f, route_i);
  }

  // 4) capacity scan
  capacity_kernel<<<1, 32, 0, stream>>>(route_i, slot_i, keep_i, dest_i);

  // 5) gather
  gather_kernel<<<NEXP * CAPP, 256, 0, stream>>>(tok_h, dest_i, xin_h);

  // 6) FFN layer 1 (TDM LDS staging + WMMA + bias + relu)
  {
    dim3 g(DFF / BN, CAPP / BM, NEXP);
    ffn1_wmma<<<g, 256, 0, stream>>>(xin_h, w1t_h, b1, h_h);
  }

  // 7) FFN layer 2 (async-copy LDS staging + WMMA + bias)
  {
    dim3 g(DMODEL / BN, CAPP / BM, NEXP);
    ffn2_wmma<<<g, 256, 0, stream>>>(h_h, w2t_h, b2, eo_f);
  }

  // 8) scatter + passthrough + *pmax + sum pool
  {
    dim3 g((DMODEL + 255) / 256, BATCH, 1);
    scatter_pool_kernel<<<g, 256, 0, stream>>>(eo_f, xt_f, route_i, slot_i, keep_i,
                                               pmax_f, pooled);
  }

  // 9) head GEMM
  head_wmma<<<NCLS_PAD / 16, 32, 0, stream>>>(pooled, hwt_h, head_b, out);
}
